// BalancedFocalLoss_66846870995520
// MI455X (gfx1250) — matrix operand order repaired and checked by reading, compiled, not run on hardware
//
#include <hip/hip_runtime.h>

typedef __attribute__((ext_vector_type(16))) _Float16 v16h;
typedef __attribute__((ext_vector_type(8)))  _Float16 v8h;
typedef __attribute__((ext_vector_type(8)))  float    v8f;
typedef __attribute__((ext_vector_type(4)))  float    v4f;

#define BSZ   4096
#define NROW  8192
#define DIM   256
#define NT    (NROW / 16)   // 512 column tiles
#define CG    4             // column groups (grid.y)
#define TPG   (NT / CG)     // 128 tiles per column group
#define LDS_H 264           // halfs per staged row: 256 data + 8 pad (bank skew)
#define BUFB  (16 * LDS_H)  // halfs per staged tile buffer

// ---------------------------------------------------------------------------
// Kernel 1: row-normalize z = [zx; zy], store f16. One wave per row.
// ---------------------------------------------------------------------------
__global__ __launch_bounds__(256) void nrm_kernel(const float* __restrict__ zx,
                                                  const float* __restrict__ zy,
                                                  _Float16* __restrict__ znh) {
  const int wave = threadIdx.x >> 5, lane = threadIdx.x & 31;
  const int row  = blockIdx.x * 8 + wave;
  const float* src = (row < BSZ) ? (zx + (size_t)row * DIM)
                                 : (zy + (size_t)(row - BSZ) * DIM);
  v4f a = *(const v4f*)(src + lane * 8);
  v4f b = *(const v4f*)(src + lane * 8 + 4);
  float ss = 0.f;
#pragma unroll
  for (int e = 0; e < 4; ++e) ss += a[e] * a[e] + b[e] * b[e];
#pragma unroll
  for (int m = 16; m >= 1; m >>= 1) ss += __shfl_xor(ss, m, 32);
  const float scl = rsqrtf(ss);
  v8h o;
#pragma unroll
  for (int e = 0; e < 4; ++e) {
    o[e]     = (_Float16)(a[e] * scl);
    o[e + 4] = (_Float16)(b[e] * scl);
  }
  *(v8h*)(znh + (size_t)row * DIM + lane * 8) = o;
}

// ---------------------------------------------------------------------------
// Kernel 2: positive logit per row: pos[i] = 2 * dot(zn[i], zn[i^BSZ]).
// ---------------------------------------------------------------------------
__global__ __launch_bounds__(256) void pos_kernel(const _Float16* __restrict__ znh,
                                                  float* __restrict__ pos) {
  const int wave = threadIdx.x >> 5, lane = threadIdx.x & 31;
  const int row  = blockIdx.x * 8 + wave;
  v8h a = *(const v8h*)(znh + (size_t)row * DIM + lane * 8);
  v8h b = *(const v8h*)(znh + (size_t)(row ^ BSZ) * DIM + lane * 8);
  float s = 0.f;
#pragma unroll
  for (int e = 0; e < 8; ++e) s += (float)a[e] * (float)b[e];
#pragma unroll
  for (int m = 16; m >= 1; m >>= 1) s += __shfl_xor(s, m, 32);
  if (lane == 0) pos[row] = 2.0f * s;
}

// ---------------------------------------------------------------------------
// Kernel 3: Gram pass. Block = 8 waves, each owning its own 16-row A tile
// (128 rows/block, A resident in VGPRs). All waves share each 16-col B tile,
// staged via async global->LDS b128 copies (ASYNCcnt), TRIPLE-buffered so a
// single barrier per tile suffices: stage(t+1) overwrites the buffer last
// read at t-2, ordered by the lead barrier of t-1.
// rowpart[g][row] = sum_{j in slice g, j != row} exp(sim[row][j]).
// ---------------------------------------------------------------------------
__global__ __launch_bounds__(256) void gram_kernel(const _Float16* __restrict__ znh,
                                                   float* __restrict__ rowpart) {
  __shared__ _Float16 sB[3][BUFB];

  const int I0    = blockIdx.x * 128;
  const int g     = blockIdx.y;
  const int wave  = threadIdx.x >> 5, lane = threadIdx.x & 31;
  const int laneM = lane & 15, laneH = lane >> 4;

  // Preload this wave's 8 A fragments (full K=256, ISA 7.12.2 layout).
  const _Float16* rowA = znh + (size_t)(I0 + wave * 16 + laneM) * DIM + laneH * 8;
  union AF { v16h v; v8h h[2]; };
  AF afrag[8];
#pragma unroll
  for (int s = 0; s < 8; ++s) {
    afrag[s].h[0] = *(const v8h*)(rowA + 32 * s);
    afrag[s].h[1] = *(const v8h*)(rowA + 32 * s + 16);
  }

  const unsigned ldsb0 = (unsigned)(uintptr_t)&sB[0][0];

  // Stage one 16x256 f16 B tile (8 KB): 512 16-byte chunks, 2 async b128
  // copies per thread. Rows padded to LDS_H halfs for bank skew.
  auto stage = [&](int jt, int buf) {
    const unsigned base = ldsb0 + (unsigned)buf * (BUFB * 2);
#pragma unroll
    for (int k = 0; k < 2; ++k) {
      const int c = threadIdx.x + k * 256;          // chunk id 0..511
      const int r = c >> 5;                         // row in tile (32 chunks/row)
      const int o = (c & 31) * 16;                  // byte offset within row
      const unsigned long long gsrc =
          (unsigned long long)(const void*)((const char*)(znh + (size_t)(jt * 16 + r) * DIM) + o);
      const unsigned ldst = base + (unsigned)(r * (LDS_H * 2) + o);
      asm volatile("global_load_async_to_lds_b128 %0, %1, off"
                   :: "v"(ldst), "v"(gsrc) : "memory");
    }
  };

  float part[8];
#pragma unroll
  for (int r = 0; r < 8; ++r) part[r] = 0.f;

  const int jt0 = g * TPG;
  stage(jt0, 0);
  int cur = 0, nxt = 1;
  for (int t = 0; t < TPG; ++t) {
    if (t + 1 < TPG) {
      stage(jt0 + t + 1, nxt);
      asm volatile("s_wait_asynccnt 0x2" ::: "memory");  // tile t's 2 copies done
    } else {
      asm volatile("s_wait_asynccnt 0x0" ::: "memory");
    }
    __syncthreads();   // all waves' copies for tile t visible (single barrier/tile)

    // B fragment: two 8-half chunks per lane (k = 8h..8h+7, 16+8h..16+8h+7) from the staged tile.
    const _Float16* bb = &sB[0][0] + cur * BUFB + laneM * LDS_H + laneH * 8;
    AF bfrag[8];
#pragma unroll
    for (int s = 0; s < 8; ++s) {
      bfrag[s].h[0] = *(const v8h*)(bb + 32 * s);
      bfrag[s].h[1] = *(const v8h*)(bb + 32 * s + 16);
    }
    v8f c = {};
#pragma unroll
    for (int s = 0; s < 8; ++s)
      c = __builtin_amdgcn_wmma_f32_16x16x32_f16(false, afrag[s].v, false, bfrag[s].v,
                                                 (short)0, c, false, false);

    const int col = (jt0 + t) * 16 + laneM;
#pragma unroll
    for (int r = 0; r < 8; ++r) {
      const int row = I0 + wave * 16 + laneH * 8 + r;
      const float e = __expf(c[r] * 2.0f);          // sim = dot / TEMPERATURE
      part[r] += (col != row) ? e : 0.0f;           // mask diagonal (cndmask)
    }

    cur = nxt;
    nxt = (nxt == 2) ? 0 : nxt + 1;
  }

  // Reduce over the 16 lanes sharing each row (per half-wave group).
#pragma unroll
  for (int r = 0; r < 8; ++r) {
#pragma unroll
    for (int m = 8; m >= 1; m >>= 1) part[r] += __shfl_xor(part[r], m, 16);
  }
  if (laneM == 0) {
    v8f pv;
#pragma unroll
    for (int r = 0; r < 8; ++r) pv[r] = part[r];
    *(v8f*)(rowpart + (size_t)g * NROW + I0 + wave * 16 + laneH * 8) = pv;
  }
}

// ---------------------------------------------------------------------------
// Kernel 4: per-row combine + focal transform; fixed-order block reduction.
// ---------------------------------------------------------------------------
__global__ __launch_bounds__(256) void finish_kernel(const float* __restrict__ rowpart,
                                                     const float* __restrict__ pos,
                                                     float* __restrict__ blockpart) {
  __shared__ float s_f[256];
  const int row = blockIdx.x * 256 + threadIdx.x;
  float s = 0.f;
#pragma unroll
  for (int gp = 0; gp < CG; ++gp) s += rowpart[(size_t)gp * NROW + row];
  const float ce = __logf(s) - pos[row];
  const float pt = __expf(-ce);
  const float om = 1.0f - pt;
  s_f[threadIdx.x] = om * om * ce;                  // GAMMA = 2
  __syncthreads();
  if (threadIdx.x == 0) {
    float t = 0.f;
    for (int i = 0; i < 256; ++i) t += s_f[i];
    blockpart[blockIdx.x] = t;
  }
}

// ---------------------------------------------------------------------------
// Kernel 5: deterministic fixed-order final sum.
// ---------------------------------------------------------------------------
__global__ void final_kernel(const float* __restrict__ blockpart,
                             float* __restrict__ out) {
  if (threadIdx.x == 0 && blockIdx.x == 0) {
    float t = 0.f;
    for (int i = 0; i < NROW / 256; ++i) t += blockpart[i];
    out[0] = 0.25f * t / (float)NROW;               // ALPHA * mean
  }
}

extern "C" void kernel_launch(void* const* d_in, const int* in_sizes, int n_in,
                              void* d_out, int out_size, void* d_ws, size_t ws_size,
                              hipStream_t stream) {
  const float* zx = (const float*)d_in[0];
  const float* zy = (const float*)d_in[1];

  char* ws = (char*)d_ws;
  _Float16* znh    = (_Float16*)ws;                                  // 4 MB
  float* pos       = (float*)(ws + (size_t)NROW * DIM * 2);          // 32 KB
  float* rowpart   = (float*)(ws + (size_t)NROW * DIM * 2 + NROW * 4);          // 128 KB
  float* blockpart = (float*)(ws + (size_t)NROW * DIM * 2 + NROW * 4 + (size_t)CG * NROW * 4);

  nrm_kernel<<<NROW / 8, 256, 0, stream>>>(zx, zy, znh);
  pos_kernel<<<NROW / 8, 256, 0, stream>>>(znh, pos);
  gram_kernel<<<dim3(NROW / 128, CG), 256, 0, stream>>>(znh, rowpart);
  finish_kernel<<<NROW / 256, 256, 0, stream>>>(rowpart, pos, blockpart);
  final_kernel<<<1, 64, 0, stream>>>(blockpart, (float*)d_out);
}
